// GCN_Brain_18081812316376
// MI455X (gfx1250) — matrix operand order repaired and checked by reading, not compiled, not run on hardware
//
#include <hip/hip_runtime.h>
#include <hip/hip_bf16.h>

typedef __attribute__((ext_vector_type(16))) _Float16 v16h;
typedef __attribute__((ext_vector_type(8)))  _Float16 v8h;
typedef __attribute__((ext_vector_type(8)))  float    v8f;

// ---------------------------------------------------------------------------
// Small elementwise kernels
// ---------------------------------------------------------------------------

// Convert W[128,128] f32 row-major -> Wt[128,128] f16 TRANSPOSED (Wt[c][k])
__global__ void k_w_to_f16_t(const float* __restrict__ W,
                             _Float16* __restrict__ Wt, int n) {
  int i = blockIdx.x * blockDim.x + threadIdx.x;
  if (i >= n) return;
  int k = i >> 7, c = i & 127;
  Wt[c * 128 + k] = (_Float16)W[i];
}

// h0[n, 0:64] = nan_to_num(x), h0[n, 64:128] = isnan(x)
__global__ void k_build_h0(const float* __restrict__ x,
                           _Float16* __restrict__ h, int N) {
  int i = blockIdx.x * blockDim.x + threadIdx.x;
  if (i >= N * 64) return;
  int n = i >> 6, f = i & 63;
  float v = x[i];
  bool nan = (v != v);
  h[(size_t)n * 128 + f]      = (_Float16)(nan ? 0.0f : v);
  h[(size_t)n * 128 + 64 + f] = (_Float16)(nan ? 1.0f : 0.0f);
}

// wabs[e] = |nan_to_num(ea[e])| ; deg[col[e]] += wabs[e]
__global__ void k_edge_deg(const int* __restrict__ col,
                           const float* __restrict__ ea,
                           float* __restrict__ deg,
                           float* __restrict__ wabs, int E) {
  int e = blockIdx.x * blockDim.x + threadIdx.x;
  if (e >= E) return;
  float w = ea[e];
  w = (w != w) ? 0.0f : fabsf(w);
  wabs[e] = w;
  atomicAdd(&deg[col[e]], w);
}

// dinv = rsqrt(deg + 1)   (deg+1 >= 1 always, so no zero branch needed)
__global__ void k_dinv(const float* __restrict__ deg,
                       float* __restrict__ dinv, int N) {
  int n = blockIdx.x * blockDim.x + threadIdx.x;
  if (n < N) dinv[n] = rsqrtf(deg[n] + 1.0f);
}

__global__ void k_count(const int* __restrict__ batch,
                        float* __restrict__ cnt, int N) {
  int n = blockIdx.x * blockDim.x + threadIdx.x;
  if (n < N) atomicAdd(&cnt[batch[n]], 1.0f);
}

// ---------------------------------------------------------------------------
// WMMA GEMM: C[N,128] = A[N,128](f16) x W[128,128](f16), f32 accumulate.
// W is provided TRANSPOSED (Wt[c][k]) so every B-fragment slice is a
// contiguous 16-half run -> ds_load_b128 pairs, no strided u16 gathers.
// Block = 256 threads = 8 waves; wave j owns output columns [16j,16j+16);
// each block processes ROWT row-tiles (8*16 = 128 rows) per 32KB LDS stage.
// LDS pitch padded to 136 halves (68 dwords) -> near-conflict-free banks.
// ---------------------------------------------------------------------------
#define ROWT 8

__global__ void __launch_bounds__(256)
k_gemm_f16(const _Float16* __restrict__ A, const _Float16* __restrict__ Wt,
           float* __restrict__ C, int ntiles) {
  __shared__ _Float16 lw[128 * 136];  // 34 KB, padded pitch
  // Stage Wt into LDS (row c at lw + c*136), uint4 (8-half) chunks
  for (int i = threadIdx.x; i < 128 * 16; i += 256) {
    int r = i >> 4, q = i & 15;
    ((uint4*)(lw + r * 136))[q] = ((const uint4*)(Wt + r * 128))[q];
  }
  __syncthreads();

  const int wave = threadIdx.x >> 5;   // 0..7 -> column tile
  const int lane = threadIdx.x & 31;
  const int m    = lane & 15;          // A-row within tile / B-column
  const int hsel = lane >> 4;          // half-select per ISA layouts

  // B fragments are invariant across row tiles: preload all 4 K-steps.
  const _Float16* bbase = lw + (wave * 16 + m) * 136;
  v16h bf[4];
  #pragma unroll
  for (int ks = 0; ks < 4; ++ks) {
    v8h blo = *(const v8h*)(bbase + ks * 32 + hsel * 16);
    v8h bhi = *(const v8h*)(bbase + ks * 32 + hsel * 16 + 8);
    bf[ks] = __builtin_shufflevector(blo, bhi,
             0,1,2,3,4,5,6,7,8,9,10,11,12,13,14,15);
  }

  const int t0 = blockIdx.x * ROWT;
  const int t1 = (t0 + ROWT < ntiles) ? (t0 + ROWT) : ntiles;
  for (int t = t0; t < t1; ++t) {          // uniform bound: EXEC stays ~0xFFFFFFFF
    const _Float16* ap = A + ((size_t)t * 16 + m) * 128;
    v8f acc = {};
    #pragma unroll
    for (int ks = 0; ks < 4; ++ks) {
      // A k order used here (contiguous): lanes 0-15 K{0-15}, lanes 16-31 K{16-31} (matches B)
      v8h alo = *(const v8h*)(ap + ks * 32 + hsel * 16);
      v8h ahi = *(const v8h*)(ap + ks * 32 + hsel * 16 + 8);
      v16h a = __builtin_shufflevector(alo, ahi,
               0,1,2,3,4,5,6,7,8,9,10,11,12,13,14,15);
      acc = __builtin_amdgcn_wmma_f32_16x16x32_f16(
          /*neg_a=*/false, a, /*neg_b=*/false, bf[ks],
          /*c_mod=*/(short)0, acc, /*reuse_a=*/false, /*reuse_b=*/false);
    }
    // C/D layout: VGPR i -> M = i + 8*(lane>>4), N = lane&15
    const int moff = hsel * 8;
    #pragma unroll
    for (int i = 0; i < 8; ++i)
      C[((size_t)t * 16 + moff + i) * 128 + wave * 16 + m] = acc[i];
  }
}

// ---------------------------------------------------------------------------
// Edge aggregation: agg[col] += dinv[row]*w*dinv[col] * hW[row]
// One wave per edge, 8 edges per block, float4 per lane (32*4 = 128 feats).
// ---------------------------------------------------------------------------
__global__ void __launch_bounds__(256)
k_edge_agg(const int* __restrict__ row, const int* __restrict__ col,
           const float* __restrict__ wabs, const float* __restrict__ dinv,
           const float* __restrict__ hW, float* __restrict__ agg, int E) {
  int e = blockIdx.x * 8 + (threadIdx.x >> 5);
  if (e >= E) return;
  int lane = threadIdx.x & 31;
  int r = row[e], c = col[e];
  float norm = dinv[r] * wabs[e] * dinv[c];
  float4 v = ((const float4*)(hW + (size_t)r * 128))[lane];
  float* dst = agg + (size_t)c * 128 + lane * 4;
  atomicAdd(dst + 0, norm * v.x);
  atomicAdd(dst + 1, norm * v.y);
  atomicAdd(dst + 2, norm * v.z);
  atomicAdd(dst + 3, norm * v.w);
}

// ---------------------------------------------------------------------------
// Post-ops, layers 0/1: self-loop + bias + BN + ReLU -> f16 for next layer
// ---------------------------------------------------------------------------
__global__ void k_post_bnrelu(const float* __restrict__ agg,
                              const float* __restrict__ hW,
                              const float* __restrict__ dinv,
                              const float* __restrict__ bias,
                              const float* __restrict__ gamma,
                              const float* __restrict__ beta,
                              const float* __restrict__ mean,
                              const float* __restrict__ var,
                              _Float16* __restrict__ hout, int N) {
  int i = blockIdx.x * blockDim.x + threadIdx.x;
  if (i >= N * 128) return;
  int n = i >> 7, f = i & 127;
  float di = dinv[n];
  float v = agg[i] + di * di * hW[i] + bias[f];
  v = (v - mean[f]) * rsqrtf(var[f] + 1e-5f) * gamma[f] + beta[f];
  hout[i] = (_Float16)fmaxf(v, 0.0f);
}

// Post-op, layer 2: self-loop + bias, then scatter into per-graph pool sums
__global__ void k_post_pool(const float* __restrict__ agg,
                            const float* __restrict__ hW,
                            const float* __restrict__ dinv,
                            const float* __restrict__ bias,
                            const int* __restrict__ batch,
                            float* __restrict__ pool, int N) {
  int i = blockIdx.x * blockDim.x + threadIdx.x;
  if (i >= N * 128) return;
  int n = i >> 7, f = i & 127;
  float di = dinv[n];
  float v = agg[i] + di * di * hW[i] + bias[f];
  atomicAdd(&pool[(size_t)batch[n] * 128 + f], v);
}

// ---------------------------------------------------------------------------
// Head MLP: g = pool/cnt; t = GELU(g@mW1+mb1); out = t@mW2+mb2  (64x128->64x16)
// Single block; trivial FLOPs, fp32 for fidelity. Exact-erf GELU.
// ---------------------------------------------------------------------------
__global__ void __launch_bounds__(256)
k_head(const float* __restrict__ pool, const float* __restrict__ cnt,
       const float* __restrict__ mW1, const float* __restrict__ mb1,
       const float* __restrict__ mW2, const float* __restrict__ mb2,
       float* __restrict__ out) {
  __shared__ float g[64 * 128];
  __shared__ float t[64 * 128];
  for (int i = threadIdx.x; i < 64 * 128; i += 256) {
    int gi = i >> 7;
    g[i] = pool[i] / fmaxf(cnt[gi], 1.0f);
  }
  __syncthreads();
  for (int i = threadIdx.x; i < 64 * 128; i += 256) {
    int gi = i >> 7, hj = i & 127;
    float s = mb1[hj];
    #pragma unroll 4
    for (int k = 0; k < 128; ++k) s += g[gi * 128 + k] * mW1[k * 128 + hj];
    t[i] = 0.5f * s * (1.0f + erff(s * 0.70710678118654752f));
  }
  __syncthreads();
  for (int i = threadIdx.x; i < 64 * 16; i += 256) {
    int gi = i >> 4, oj = i & 15;
    float s = mb2[oj];
    #pragma unroll 4
    for (int k = 0; k < 128; ++k) s += t[gi * 128 + k] * mW2[k * 16 + oj];
    out[i] = s;
  }
}

// ---------------------------------------------------------------------------
// Host launcher
// ---------------------------------------------------------------------------
extern "C" void kernel_launch(void* const* d_in, const int* in_sizes, int n_in,
                              void* d_out, int out_size, void* d_ws, size_t ws_size,
                              hipStream_t stream) {
  const float* x     = (const float*)d_in[0];
  const int*   ei    = (const int*)  d_in[1];   // [2, E] flat: row then col
  const float* ea    = (const float*)d_in[2];
  const int*   batch = (const int*)  d_in[3];
  const float* W0    = (const float*)d_in[4];
  const float* b0    = (const float*)d_in[5];
  const float* W1    = (const float*)d_in[6];
  const float* b1    = (const float*)d_in[7];
  const float* W2    = (const float*)d_in[8];
  const float* b2    = (const float*)d_in[9];
  const float* bn_g  = (const float*)d_in[10];
  const float* bn_b  = (const float*)d_in[11];
  const float* bn_m  = (const float*)d_in[12];
  const float* bn_v  = (const float*)d_in[13];
  const float* mW1   = (const float*)d_in[14];
  const float* mb1   = (const float*)d_in[15];
  const float* mW2   = (const float*)d_in[16];
  const float* mb2   = (const float*)d_in[17];

  const int N = in_sizes[0] / 64;   // 50000 (multiple of 16)
  const int E = in_sizes[2];        // 800000
  const int* erow = ei;
  const int* ecol = ei + E;

  // Workspace carve-out (256B aligned slices)
  char* ws = (char*)d_ws;
  size_t off = 0;
  auto carve = [&](size_t bytes) -> void* {
    void* p = ws + off;
    off = (off + bytes + 255) & ~(size_t)255;
    return p;
  };
  _Float16* hf   = (_Float16*)carve((size_t)N * 128 * sizeof(_Float16));
  float*    hW   = (float*)   carve((size_t)N * 128 * sizeof(float));
  float*    agg  = (float*)   carve((size_t)N * 128 * sizeof(float));
  float*    deg  = (float*)   carve((size_t)N * sizeof(float));
  float*    dinv = (float*)   carve((size_t)N * sizeof(float));
  float*    wabs = (float*)   carve((size_t)E * sizeof(float));
  _Float16* w16  = (_Float16*)carve((size_t)3 * 128 * 128 * sizeof(_Float16));
  float*    pool = (float*)   carve((size_t)64 * 128 * sizeof(float));
  float*    cnt  = (float*)   carve((size_t)64 * sizeof(float));
  (void)ws_size; (void)n_in; (void)out_size;

  const int T = 256;

  // Zero accumulators (graph-capture-safe memset nodes)
  hipMemsetAsync(deg,  0, (size_t)N * sizeof(float), stream);
  hipMemsetAsync(pool, 0, (size_t)64 * 128 * sizeof(float), stream);
  hipMemsetAsync(cnt,  0, (size_t)64 * sizeof(float), stream);

  // Weight conversion (transposed) + input build + degree/dinv + counts
  k_w_to_f16_t<<<(16384 + T - 1) / T, T, 0, stream>>>(W0, w16 + 0 * 16384, 16384);
  k_w_to_f16_t<<<(16384 + T - 1) / T, T, 0, stream>>>(W1, w16 + 1 * 16384, 16384);
  k_w_to_f16_t<<<(16384 + T - 1) / T, T, 0, stream>>>(W2, w16 + 2 * 16384, 16384);
  k_build_h0<<<((size_t)N * 64 + T - 1) / T, T, 0, stream>>>(x, hf, N);
  k_edge_deg<<<(E + T - 1) / T, T, 0, stream>>>(ecol, ea, deg, wabs, E);
  k_dinv<<<(N + T - 1) / T, T, 0, stream>>>(deg, dinv, N);
  k_count<<<(N + T - 1) / T, T, 0, stream>>>(batch, cnt, N);

  const int ntiles      = N / 16;                       // 3125
  const int gemm_blocks = (ntiles + ROWT - 1) / ROWT;   // 391
  const int post_blocks = ((size_t)N * 128 + T - 1) / T;
  const int edge_blocks = (E + 7) / 8;

  // ---- Layer 0 ----
  hipMemsetAsync(agg, 0, (size_t)N * 128 * sizeof(float), stream);
  k_gemm_f16<<<gemm_blocks, T, 0, stream>>>(hf, w16 + 0 * 16384, hW, ntiles);
  k_edge_agg<<<edge_blocks, T, 0, stream>>>(erow, ecol, wabs, dinv, hW, agg, E);
  k_post_bnrelu<<<post_blocks, T, 0, stream>>>(agg, hW, dinv, b0,
                                               bn_g, bn_b, bn_m, bn_v, hf, N);
  // ---- Layer 1 ----
  hipMemsetAsync(agg, 0, (size_t)N * 128 * sizeof(float), stream);
  k_gemm_f16<<<gemm_blocks, T, 0, stream>>>(hf, w16 + 1 * 16384, hW, ntiles);
  k_edge_agg<<<edge_blocks, T, 0, stream>>>(erow, ecol, wabs, dinv, hW, agg, E);
  k_post_bnrelu<<<post_blocks, T, 0, stream>>>(agg, hW, dinv, b1,
                                               bn_g, bn_b, bn_m, bn_v, hf, N);
  // ---- Layer 2 (no BN/ReLU) + global mean pool ----
  hipMemsetAsync(agg, 0, (size_t)N * 128 * sizeof(float), stream);
  k_gemm_f16<<<gemm_blocks, T, 0, stream>>>(hf, w16 + 2 * 16384, hW, ntiles);
  k_edge_agg<<<edge_blocks, T, 0, stream>>>(erow, ecol, wabs, dinv, hW, agg, E);
  k_post_pool<<<post_blocks, T, 0, stream>>>(agg, hW, dinv, b2, batch, pool, N);

  // ---- Head MLP ----
  k_head<<<1, T, 0, stream>>>(pool, cnt, mW1, mb1, mW2, mb2, (float*)d_out);
}
